// Encoder_25915832664270
// MI455X (gfx1250) — compile-verified, matrix operand order checked by reading, not run on hardware
//
#include <hip/hip_runtime.h>
#include <math.h>

// ---------------- problem constants ----------------
#define BB 32
#define NN 2048
#define SS 64
#define KK 128
#define R2 (0.2f * 0.2f)

// ---------------- workspace layout (bytes, 256-aligned) ----------------
#define OFF_CENTERS 0u               // 2048*3 f32            = 24576
#define OFF_NBR     24576u           // 262144 i32            = 1048576
#define OFF_REL     1073152u         // 262144*3 f32          = 3145728
#define OFF_G       4218880u         // 2048*544 f16          = 2228224
#define OFF_G2      6447104u         // 2048*512 f16          = 2097152
#define OFF_W2T     8544256u         // 128*64  f16           = 16384
#define OFF_W3T     8560640u         // 512*128 f16           = 131072
#define OFF_W4T     8691712u         // 512*544 f16           = 557056
#define OFF_W5T     9248768u         // 1024*512 f16          = 1048576
// total ~10.3 MB

typedef __attribute__((ext_vector_type(16))) _Float16 v16h;
typedef __attribute__((ext_vector_type(8)))  _Float16 v8h;
typedef __attribute__((ext_vector_type(8)))  float    v8f;

__device__ __forceinline__ float leaky(float x) { return x >= 0.f ? x : 0.2f * x; }

// WMMA D = A*B + C, f16 inputs, f32 accumulate (codegen-verified builtin)
__device__ __forceinline__ v8f wmma16(v16h a, v16h b, v8f c) {
  return __builtin_amdgcn_wmma_f32_16x16x32_f16(false, a, false, b, (short)0, c,
                                                false, false);
}

// Load one 16x32 A-fragment (or B-fragment from an [N][K] transposed matrix)
// from a row-major f16 matrix. `tile` points at (row0, k0) of the 16-row tile.
// ISA 7.12.2 layout: lane = m + 16*kg; halves i=0..7 -> k=8*kg+i,
// i=8..15 -> k=16+8*kg+(i-8).  => two contiguous 16B loads per lane.
__device__ __forceinline__ v16h frag_ld(const _Float16* tile, int stride, int kOff,
                                        unsigned lane) {
  unsigned m  = lane & 15u;
  unsigned kg = (lane >> 4) & 1u;
  const _Float16* p = tile + (size_t)m * stride + kOff + 8u * kg;
  v8h lo = *(const v8h*)(p);
  v8h hi = *(const v8h*)(p + 16);
  v16h r;
#pragma unroll
  for (int i = 0; i < 8; ++i) { r[i] = lo[i]; r[i + 8] = hi[i]; }
  return r;
}

// ---------------- kernel 1: convert + transpose weights to f16 ----------------
// W2t[n][k] (128x64), W3t[n][k] (512x128), W4t[n][k] (512x544, K padded 515->544),
// W5t[n][k] (1024x512)
__global__ void prep_weights_kernel(const float* __restrict__ W2,
                                    const float* __restrict__ W3,
                                    const float* __restrict__ W4,
                                    const float* __restrict__ W5,
                                    _Float16* __restrict__ w2t,
                                    _Float16* __restrict__ w3t,
                                    _Float16* __restrict__ w4t,
                                    _Float16* __restrict__ w5t) {
  int i = blockIdx.x * blockDim.x + threadIdx.x;
  if (i < 128 * 64) {
    int n = i / 64, k = i % 64;
    w2t[i] = (_Float16)W2[k * 128 + n];
    return;
  }
  int j = i - 128 * 64;
  if (j < 512 * 128) {
    int n = j / 128, k = j % 128;
    w3t[j] = (_Float16)W3[k * 512 + n];
    return;
  }
  int l = j - 512 * 128;
  if (l < 512 * 544) {
    int n = l / 544, k = l % 544;
    w4t[l] = (k < 515) ? (_Float16)W4[k * 512 + n] : (_Float16)0.f;
    return;
  }
  int q = l - 512 * 544;
  if (q < 1024 * 512) {
    int n = q / 512, k = q % 512;
    w5t[q] = (_Float16)W5[k * 1024 + n];
  }
}

// ---------------- kernel 2: farthest point sampling -> centers ----------------
__global__ void fps_kernel(const float* __restrict__ pos, float* __restrict__ centers) {
  int b = blockIdx.x;
  int tid = threadIdx.x;  // 256 threads
  __shared__ float spx[NN], spy[NN], spz[NN];
  __shared__ float sdist[NN];
  __shared__ float rbv[256];
  __shared__ int   rbi[256];

  for (int i = tid; i < NN; i += 256) {
    const float* p = pos + (size_t)(b * NN + i) * 3;
    spx[i] = p[0]; spy[i] = p[1]; spz[i] = p[2];
  }
  __syncthreads();
  if (tid == 0) {
    centers[(b * SS + 0) * 3 + 0] = spx[0];
    centers[(b * SS + 0) * 3 + 1] = spy[0];
    centers[(b * SS + 0) * 3 + 2] = spz[0];
  }
  for (int i = tid; i < NN; i += 256) {
    float dx = spx[i] - spx[0], dy = spy[i] - spy[0], dz = spz[i] - spz[0];
    sdist[i] = dx * dx + dy * dy + dz * dz;
  }
  __syncthreads();
  for (int s = 1; s < SS; ++s) {
    // argmax (first occurrence on ties)
    float bv = -1.f; int bi = 0;
    for (int i = tid; i < NN; i += 256) {
      float v = sdist[i];
      if (v > bv) { bv = v; bi = i; }
    }
    rbv[tid] = bv; rbi[tid] = bi;
    __syncthreads();
    for (int off = 128; off >= 1; off >>= 1) {
      if (tid < off) {
        float ov = rbv[tid + off]; int oi = rbi[tid + off];
        if (ov > rbv[tid] || (ov == rbv[tid] && oi < rbi[tid])) {
          rbv[tid] = ov; rbi[tid] = oi;
        }
      }
      __syncthreads();
    }
    int nxt = rbi[0];
    if (tid == 0) {
      centers[(b * SS + s) * 3 + 0] = spx[nxt];
      centers[(b * SS + s) * 3 + 1] = spy[nxt];
      centers[(b * SS + s) * 3 + 2] = spz[nxt];
    }
    float cx = spx[nxt], cy = spy[nxt], cz = spz[nxt];
    for (int i = tid; i < NN; i += 256) {
      float dx = spx[i] - cx, dy = spy[i] - cy, dz = spz[i] - cz;
      float d2 = dx * dx + dy * dy + dz * dz;
      sdist[i] = fminf(sdist[i], d2);
    }
    __syncthreads();
  }
}

// ---------------- kernel 3: ordered ball-query (first K in-ball indices) -------
__global__ void neighbors_kernel(const float* __restrict__ pos,
                                 const float* __restrict__ centers,
                                 int* __restrict__ nbr, float* __restrict__ relw,
                                 int* __restrict__ xout, int* __restrict__ yout) {
  int bs = blockIdx.x;            // 0..2047
  int b = bs / SS;
  int tid = threadIdx.x;          // 128 threads
  __shared__ int sscan[128];
  __shared__ int scnt;
  float cx = centers[bs * 3 + 0];
  float cy = centers[bs * 3 + 1];
  float cz = centers[bs * 3 + 2];
  if (tid == 0) scnt = 0;
  __syncthreads();
  for (int base = 0; base < NN; base += 128) {
    int n = base + tid;
    const float* p = pos + (size_t)(b * NN + n) * 3;
    float px = p[0], py = p[1], pz = p[2];
    float dx = cx - px, dy = cy - py, dz = cz - pz;
    float d2 = dx * dx + dy * dy + dz * dz;
    int f = (d2 < R2) ? 1 : 0;
    sscan[tid] = f;
    __syncthreads();
    // inclusive scan (Hillis-Steele)
    for (int off = 1; off < 128; off <<= 1) {
      int add = (tid >= off) ? sscan[tid - off] : 0;
      __syncthreads();
      sscan[tid] += add;
      __syncthreads();
    }
    int slot = scnt + sscan[tid] - f;   // exclusive position
    int tot = sscan[127];
    if (f && slot < KK) {
      nbr[bs * KK + slot] = n;
      relw[(size_t)(bs * KK + slot) * 3 + 0] = px - cx;
      relw[(size_t)(bs * KK + slot) * 3 + 1] = py - cy;
      relw[(size_t)(bs * KK + slot) * 3 + 2] = pz - cz;
      xout[bs * KK + slot] = n + b * NN;
      yout[bs * KK + slot] = bs;
    }
    __syncthreads();
    if (tid == 0) scnt += tot;
    __syncthreads();
    if (scnt >= KK) break;
  }
  int c = scnt < KK ? scnt : KK;
  for (int k = c + tid; k < KK; k += 128) {
    nbr[bs * KK + k] = -1;
    relw[(size_t)(bs * KK + k) * 3 + 0] = 0.f;
    relw[(size_t)(bs * KK + k) * 3 + 1] = 0.f;
    relw[(size_t)(bs * KK + k) * 3 + 2] = 0.f;
    xout[bs * KK + k] = -1;
    yout[bs * KK + k] = -1;
  }
}

// ---------------- kernel 4: per-neighborhood MLP chain + masked max ------------
// One block (8 waves) per (b,s).  h1 via VALU (K=3), h2/h3 via WMMA.
__global__ void group_mlp_kernel(const float* __restrict__ relw,
                                 const int* __restrict__ nbr,
                                 const float* __restrict__ centers,
                                 const float* __restrict__ W1,
                                 const float* __restrict__ b1,
                                 const float* __restrict__ b2,
                                 const float* __restrict__ b3,
                                 const _Float16* __restrict__ w2t,
                                 const _Float16* __restrict__ w3t,
                                 _Float16* __restrict__ g) {
  int bs = blockIdx.x;
  int tid = threadIdx.x;
  unsigned lane = tid & 31u;
  int wave = tid >> 5;
  __shared__ __align__(16) _Float16 sh1[128 * 64];
  __shared__ __align__(16) _Float16 sh2[128 * 128];
  __shared__ float sfeat[512];
  __shared__ float srel[128 * 3];
  __shared__ int   sval[128];

  if (tid < 128) {
    int nv = nbr[bs * KK + tid];
    sval[tid] = (nv >= 0);
    srel[tid * 3 + 0] = relw[(size_t)(bs * KK + tid) * 3 + 0];
    srel[tid * 3 + 1] = relw[(size_t)(bs * KK + tid) * 3 + 1];
    srel[tid * 3 + 2] = relw[(size_t)(bs * KK + tid) * 3 + 2];
  }
  sfeat[tid] = -INFINITY;
  sfeat[tid + 256] = -INFINITY;
  __syncthreads();

  // h1 = leaky(rel @ W1 + b1) : [128,64], K=3 -> plain FMA, store f16
  for (int e = tid; e < 128 * 64; e += 256) {
    int m = e >> 6, n = e & 63;
    float v = b1[n] + srel[m * 3 + 0] * W1[n] + srel[m * 3 + 1] * W1[64 + n] +
              srel[m * 3 + 2] * W1[128 + n];
    sh1[m * 64 + n] = (_Float16)leaky(v);
  }
  __syncthreads();

  // h2 = leaky(h1 @ W2 + b2) : [128,128] via WMMA (8x8 tiles, 8 per wave)
  for (int t8 = 0; t8 < 8; ++t8) {
    int tile = wave * 8 + t8;
    int mt = tile >> 3, nt = tile & 7;
    v8f acc = {};
#pragma unroll
    for (int ks = 0; ks < 64; ks += 32) {
      v16h a = frag_ld(sh1 + mt * 16 * 64, 64, ks, lane);
      v16h bb = frag_ld(w2t + nt * 16 * 64, 64, ks, lane);
      acc = wmma16(a, bb, acc);
    }
    int n = nt * 16 + (int)(lane & 15u);
    int mbase = mt * 16 + (int)((lane >> 4) & 1u) * 8;
    float bias = b2[n];
#pragma unroll
    for (int r = 0; r < 8; ++r) {
      sh2[(mbase + r) * 128 + n] = (_Float16)leaky(acc[r] + bias);
    }
  }
  __syncthreads();

  // h3 = leaky(h2 @ W3 + b3) : [128,512]; masked max over rows -> sfeat[512]
  for (int t = 0; t < 32; ++t) {
    int tile = wave * 32 + t;
    int mt = tile >> 5, nt = tile & 31;
    v8f acc = {};
#pragma unroll
    for (int ks = 0; ks < 128; ks += 32) {
      v16h a = frag_ld(sh2 + mt * 16 * 128, 128, ks, lane);
      v16h bb = frag_ld(w3t + nt * 16 * 128, 128, ks, lane);
      acc = wmma16(a, bb, acc);
    }
    int n = nt * 16 + (int)(lane & 15u);
    int mbase = mt * 16 + (int)((lane >> 4) & 1u) * 8;
    float bias = b3[n];
    float vmax = -INFINITY;
#pragma unroll
    for (int r = 0; r < 8; ++r) {
      float v = leaky(acc[r] + bias);
      if (sval[mbase + r]) vmax = fmaxf(vmax, v);
    }
    atomicMax(&sfeat[n], vmax);
  }
  __syncthreads();

  // g row: [feat(512) | center(3) | zero pad to 544], f16
  for (int e = tid; e < 544; e += 256) {
    float v;
    if (e < 512)      v = sfeat[e];
    else if (e < 515) v = centers[bs * 3 + (e - 512)];
    else              v = 0.f;
    g[(size_t)bs * 544 + e] = (_Float16)v;
  }
}

// ---------------- kernel 5: g2 = leaky(g @ W4 + b4), [2048,544]x[544,512] -------
__global__ void gemm4_kernel(const _Float16* __restrict__ g,
                             const _Float16* __restrict__ w4t,
                             const float* __restrict__ b4,
                             _Float16* __restrict__ g2) {
  unsigned lane = threadIdx.x & 31u;
  int tile = blockIdx.x * 8 + (threadIdx.x >> 5);  // 0..4095 (128 mt x 32 nt)
  int mt = tile >> 5, nt = tile & 31;
  v8f acc = {};
  for (int ks = 0; ks < 544; ks += 32) {
    v16h a = frag_ld(g + (size_t)(mt * 16) * 544, 544, ks, lane);
    v16h b = frag_ld(w4t + (size_t)(nt * 16) * 544, 544, ks, lane);
    acc = wmma16(a, b, acc);
  }
  int n = nt * 16 + (int)(lane & 15u);
  int mbase = mt * 16 + (int)((lane >> 4) & 1u) * 8;
  float bias = b4[n];
#pragma unroll
  for (int r = 0; r < 8; ++r) {
    g2[(size_t)(mbase + r) * 512 + n] = (_Float16)leaky(acc[r] + bias);
  }
}

// ---------------- kernel 6: out = g2 @ W5 + b5; split mean / std ---------------
__global__ void gemm5_kernel(const _Float16* __restrict__ g2,
                             const _Float16* __restrict__ w5t,
                             const float* __restrict__ b5,
                             float* __restrict__ out) {
  unsigned lane = threadIdx.x & 31u;
  int tile = blockIdx.x * 8 + (threadIdx.x >> 5);  // 0..8191 (128 mt x 64 nt)
  int mt = tile >> 6, nt = tile & 63;
  v8f acc = {};
  for (int ks = 0; ks < 512; ks += 32) {
    v16h a = frag_ld(g2 + (size_t)(mt * 16) * 512, 512, ks, lane);
    v16h b = frag_ld(w5t + (size_t)(nt * 16) * 512, 512, ks, lane);
    acc = wmma16(a, b, acc);
  }
  int n = nt * 16 + (int)(lane & 15u);
  int mbase = mt * 16 + (int)((lane >> 4) & 1u) * 8;
  float bias = b5[n];
#pragma unroll
  for (int r = 0; r < 8; ++r) {
    float v = acc[r] + bias;
    int m = mbase + r;
    if (n < 512)
      out[(size_t)m * 512 + n] = v;                                   // mean
    else
      out[1048576u + (size_t)m * 512 + (n - 512)] = expf(0.5f * v);   // std
  }
}

// ---------------- launch --------------------------------------------------------
extern "C" void kernel_launch(void* const* d_in, const int* in_sizes, int n_in,
                              void* d_out, int out_size, void* d_ws, size_t ws_size,
                              hipStream_t stream) {
  (void)in_sizes; (void)n_in; (void)out_size; (void)ws_size;
  const float* pos = (const float*)d_in[0];
  // d_in[1] = batch (unused; derived analytically)
  const float* W1 = (const float*)d_in[2];
  const float* b1 = (const float*)d_in[3];
  const float* W2 = (const float*)d_in[4];
  const float* b2 = (const float*)d_in[5];
  const float* W3 = (const float*)d_in[6];
  const float* b3 = (const float*)d_in[7];
  const float* W4 = (const float*)d_in[8];
  const float* b4 = (const float*)d_in[9];
  const float* W5 = (const float*)d_in[10];
  const float* b5 = (const float*)d_in[11];

  char* ws = (char*)d_ws;
  float*     centers = (float*)(ws + OFF_CENTERS);
  int*       nbr     = (int*)(ws + OFF_NBR);
  float*     relw    = (float*)(ws + OFF_REL);
  _Float16*  g       = (_Float16*)(ws + OFF_G);
  _Float16*  g2      = (_Float16*)(ws + OFF_G2);
  _Float16*  w2t     = (_Float16*)(ws + OFF_W2T);
  _Float16*  w3t     = (_Float16*)(ws + OFF_W3T);
  _Float16*  w4t     = (_Float16*)(ws + OFF_W4T);
  _Float16*  w5t     = (_Float16*)(ws + OFF_W5T);

  float* outF = (float*)d_out;                 // mean[2048*512], std[2048*512]
  int*   xout = (int*)d_out + 2097152;         // x_idx[262144]
  int*   yout = (int*)d_out + 2359296;         // y_idx[262144]

  prep_weights_kernel<<<3424, 256, 0, stream>>>(W2, W3, W4, W5, w2t, w3t, w4t, w5t);
  fps_kernel<<<BB, 256, 0, stream>>>(pos, centers);
  neighbors_kernel<<<BB * SS, 128, 0, stream>>>(pos, centers, nbr, relw, xout, yout);
  group_mlp_kernel<<<BB * SS, 256, 0, stream>>>(relw, nbr, centers, W1, b1, b2, b3,
                                                w2t, w3t, g);
  gemm4_kernel<<<512, 256, 0, stream>>>(g, w4t, b4, g2);
  gemm5_kernel<<<1024, 256, 0, stream>>>(g2, w5t, b5, outF);
}
